// SelfAttentionHead_111669149928
// MI455X (gfx1250) — matrix operand order repaired and checked by reading, not compiled, not run on hardware
//
#include <hip/hip_runtime.h>
#include <hip/hip_bf16.h>
#include <stdint.h>

#define DEVINL __device__ __forceinline__

#if defined(__has_builtin)
#  if __has_builtin(__builtin_amdgcn_global_load_async_to_lds_b128) && \
      __has_builtin(__builtin_amdgcn_s_wait_asynccnt)
#    define HAS_ASYNC_LDS 1
#  endif
#endif
#ifndef HAS_ASYNC_LDS
#  define HAS_ASYNC_LDS 0
#endif

constexpr int BB  = 4;
constexpr int TT  = 4096;
constexpr int DIN = 1024;
constexpr int DH  = 64;

typedef __attribute__((ext_vector_type(16))) __bf16        v16bf;
typedef __attribute__((ext_vector_type(8)))  float         v8f;
typedef __attribute__((ext_vector_type(4)))  float         v4f;
typedef __attribute__((ext_vector_type(4)))  unsigned int  v4u;
typedef __attribute__((ext_vector_type(4)))  int           v4i;

#if HAS_ASYNC_LDS
typedef __attribute__((address_space(1))) v4i as1_v4i;  // global-AS vector pointer
typedef __attribute__((address_space(3))) v4i as3_v4i;  // LDS-AS vector pointer
#endif

union Frag16 {
    v16bf          v;      // WMMA operand (16 bf16 = 8 VGPRs)
    unsigned short us[16];
    v4u            q[2];   // two 16B chunks for b128 loads
};

DEVINL unsigned short f32_bf16(float f) {
    unsigned int u = __builtin_bit_cast(unsigned int, f);
    u += 0x7fffu + ((u >> 16) & 1u);   // round-to-nearest-even
    return (unsigned short)(u >> 16);
}

DEVINL v8f wmma_bf16(const Frag16& a, const Frag16& b, v8f c) {
    return __builtin_amdgcn_wmma_f32_16x16x32_bf16(
        /*neg_a=*/false, a.v, /*neg_b=*/false, b.v,
        /*c_mod=*/(short)0, c, /*reuse_a=*/false, /*reuse_b=*/false);
}

// ---------------------------------------------------------------------------
// Kernel 1: Q = (x Wq)/8, K = x Wk, Vt = (x Wv)^T   (all bf16 into workspace)
// 128 blocks x 256 threads (8 waves); each wave owns 16 output rows.
// ---------------------------------------------------------------------------
__global__ void __launch_bounds__(256, 1)
qkv_kernel(const float* __restrict__ x,
           const float* __restrict__ Wq,
           const float* __restrict__ Wk,
           const float* __restrict__ Wv,
           unsigned short* __restrict__ Qo,
           unsigned short* __restrict__ Ko,
           unsigned short* __restrict__ Vt)
{
    __shared__ __align__(16) unsigned short sWt[3][DH][32]; // [mat][n][k] bf16, 12 KB

    const int tid  = threadIdx.x;
    const int wave = tid >> 5;
    const int lane = tid & 31;
    const int lo   = lane & 15;
    const int half = lane >> 4;
    const int r0   = blockIdx.x * 128 + wave * 16;  // wave's first row
    const int row  = r0 + lo;                       // A-fragment row (M = lane%16)

    const float* Wm[3] = { Wq, Wk, Wv };

    const v8f vzero = {0.f,0.f,0.f,0.f,0.f,0.f,0.f,0.f};
    v8f acc[3][4];
    #pragma unroll
    for (int m = 0; m < 3; ++m)
        #pragma unroll
        for (int n = 0; n < 4; ++n) acc[m][n] = vzero;

    for (int kc = 0; kc < DIN / 32; ++kc) {
        __syncthreads();
        // Stage 32x64 chunk of each W into LDS, bf16, transposed to [n][k]
        for (int i = tid; i < 3 * 32 * DH; i += 256) {
            int m = i / (32 * DH);
            int r = i % (32 * DH);
            int k = r / DH;
            int n = r % DH;
            sWt[m][n][k] = f32_bf16(Wm[m][(size_t)(kc * 32 + k) * DH + n]);
        }
        __syncthreads();

        // A fragment from x: 16x32 bf16. e<8: K=half*8+e ; e>=8: K=16+half*8+(e-8)
        Frag16 a;
        const float* xp  = x + (size_t)row * DIN + kc * 32;
        const v4f*   xv0 = reinterpret_cast<const v4f*>(xp + half * 8);
        const v4f*   xv1 = reinterpret_cast<const v4f*>(xp + 16 + half * 8);
        v4f f[4];
        f[0] = xv0[0]; f[1] = xv0[1]; f[2] = xv1[0]; f[3] = xv1[1];
        #pragma unroll
        for (int e = 0; e < 16; ++e) a.us[e] = f32_bf16(f[e >> 2][e & 3]);

        #pragma unroll
        for (int m = 0; m < 3; ++m) {
            #pragma unroll
            for (int n = 0; n < 4; ++n) {
                // B fragment 32x16: e<8: kdim = half*8+e ; e>=8: kdim = 16+half*8+(e-8), N = n*16+lo
                Frag16 b;
                const v4u* wp = reinterpret_cast<const v4u*>(&sWt[m][n * 16 + lo][0]);
                b.q[0] = wp[half];
                b.q[1] = wp[2 + half];
                acc[m][n] = wmma_bf16(a, b, acc[m][n]);
            }
        }
    }

    // C layout: VGPR v holds row M = 8*half + v, col N = lo (per 16-col tile)
    #pragma unroll
    for (int n = 0; n < 4; ++n) {
        #pragma unroll
        for (int v = 0; v < 8; ++v) {
            const int mrow = r0 + 8 * half + v;
            const int hcol = n * 16 + lo;
            Qo[(size_t)mrow * DH + hcol] = f32_bf16(acc[0][n][v] * 0.125f); // /sqrt(64)
            Ko[(size_t)mrow * DH + hcol] = f32_bf16(acc[1][n][v]);
            const int bt = mrow >> 12;          // / TT
            const int t  = mrow & (TT - 1);
            Vt[((size_t)(bt * DH) + hcol) * TT + t] = f32_bf16(acc[2][n][v]);
        }
    }
}

// ---------------------------------------------------------------------------
// Async staging of one 32-key K/V tile into per-wave LDS buffers (16 x b128).
// K tile: 32 keys x 64 dims bf16 -> 4 KB, fully contiguous in global memory.
// V tile: from transposed Vt [hcol][t]; 64 rows x 64 B, global row stride 8 KB.
// Offsets are baked into the addresses (builtin's offset arg must be an ICE).
// ---------------------------------------------------------------------------
#if HAS_ASYNC_LDS
DEVINL void issue_kv_tile(const unsigned short* __restrict__ Kb,
                          const unsigned short* __restrict__ Vt,
                          int b, int jt,
                          unsigned short* kbuf, unsigned short* vbuf, int lane)
{
    const char* gk = (const char*)(Kb + (size_t)(b * TT + jt * 32) * DH) + lane * 16;
    char*       lk = (char*)kbuf + lane * 16;
    #pragma unroll
    for (int i = 0; i < 8; ++i)
        __builtin_amdgcn_global_load_async_to_lds_b128(
            (as1_v4i*)(gk + i * 512), (as3_v4i*)(lk + i * 512), 0, 0);

    const char* gv = (const char*)(Vt + (size_t)b * DH * TT + jt * 32)
                   + (size_t)(lane >> 2) * (TT * 2) + (size_t)(lane & 3) * 16;
    char*       lv = (char*)vbuf + lane * 16;
    #pragma unroll
    for (int i = 0; i < 8; ++i)
        __builtin_amdgcn_global_load_async_to_lds_b128(
            (as1_v4i*)(gv + (size_t)i * 8 * TT * 2), (as3_v4i*)(lv + i * 512), 0, 0);
}
#endif

// ---------------------------------------------------------------------------
// Kernel 2: flash attention, causal. 128 blocks x 8 waves; wave = 16 q-rows.
// ---------------------------------------------------------------------------
__global__ void __launch_bounds__(256, 1)
attn_kernel(const unsigned short* __restrict__ Q,
            const unsigned short* __restrict__ K,
            const unsigned short* __restrict__ Vt,
            float* __restrict__ out)
{
#if HAS_ASYNC_LDS
    __shared__ __align__(16) unsigned short sK[2][8][32 * DH]; // 64 KB double-buffered K tiles
    __shared__ __align__(16) unsigned short sV[2][8][DH * 32]; // 64 KB double-buffered V tiles
#endif
    __shared__ __align__(16) unsigned short sP[8][16][32];     // per-wave P tile, 8 KB

    const int tid  = threadIdx.x;
    const int wave = tid >> 5;
    const int lane = tid & 31;
    const int lo   = lane & 15;
    const int half = lane >> 4;
    const int b    = blockIdx.x >> 5;
    const int qblk = blockIdx.x & 31;
    const int q0   = qblk * 128 + wave * 16;

    // Q fragments (two 32-wide K-dim chunks), held for the whole key loop
    Frag16 qf[2];
    {
        const v4u* qp = reinterpret_cast<const v4u*>(Q + (size_t)(b * TT + q0 + lo) * DH);
        #pragma unroll
        for (int kc = 0; kc < 2; ++kc) {
            qf[kc].q[0] = qp[(kc * 32 + half * 8) >> 3];
            qf[kc].q[1] = qp[(kc * 32 + 16 + half * 8) >> 3];
        }
    }

    const v8f vzero = {0.f,0.f,0.f,0.f,0.f,0.f,0.f,0.f};
    v8f oacc[4];
    #pragma unroll
    for (int n = 0; n < 4; ++n) oacc[n] = vzero;
    float mrun[8], lrun[8];
    #pragma unroll
    for (int v = 0; v < 8; ++v) { mrun[v] = -INFINITY; lrun[v] = 0.f; }

    const int jt_last = (q0 + 15) >> 5;

#if HAS_ASYNC_LDS
    issue_kv_tile(K, Vt, b, 0, sK[0][wave], sV[0][wave], lane);  // prime the pipe
#endif

    for (int jt = 0; jt <= jt_last; ++jt) {
#if HAS_ASYNC_LDS
        const unsigned short* kbuf = sK[jt & 1][wave];
        const unsigned short* vbuf = sV[jt & 1][wave];
        if (jt < jt_last) {   // prefetch next tile while computing this one
            issue_kv_tile(K, Vt, b, jt + 1, sK[(jt + 1) & 1][wave], sV[(jt + 1) & 1][wave], lane);
            __builtin_amdgcn_s_wait_asynccnt(16);  // loads complete in order: current tile ready
        } else {
            __builtin_amdgcn_s_wait_asynccnt(0);
        }
        asm volatile("" ::: "memory");
#else
        if (jt < jt_last)  // pull next K tile toward L2/L0
            __builtin_prefetch(K + (size_t)(b * TT + (jt + 1) * 32) * DH, 0, 0);
#endif

        // ---- S = Q K^T, 16x32 as two 16x16 accumulators ----
        v8f sacc[2] = { vzero, vzero };
        #pragma unroll
        for (int n = 0; n < 2; ++n) {
#if HAS_ASYNC_LDS
            const v4u* kp = reinterpret_cast<const v4u*>(kbuf + (n * 16 + lo) * DH);
#else
            const int key = jt * 32 + n * 16 + lo;          // B-frag N = lo
            const v4u* kp = reinterpret_cast<const v4u*>(K + (size_t)(b * TT + key) * DH);
#endif
            #pragma unroll
            for (int kc = 0; kc < 2; ++kc) {
                Frag16 kf;  // e<8: kdim = kc*32 + half*8 + e ; e>=8: kdim = kc*32 + 16 + half*8 + (e-8)
                kf.q[0] = kp[(kc * 32 + half * 8) >> 3];
                kf.q[1] = kp[(kc * 32 + 16 + half * 8) >> 3];
                sacc[n] = wmma_bf16(qf[kc], kf, sacc[n]);
            }
        }

        // ---- causal mask + online softmax (row M = 8*half + v, col N = lo) ----
        #pragma unroll
        for (int v = 0; v < 8; ++v) {
            const int rrow = q0 + 8 * half + v;
            const int k0   = jt * 32 + lo;
            const float s0 = (k0      <= rrow) ? sacc[0][v] : -INFINITY;
            const float s1 = (k0 + 16 <= rrow) ? sacc[1][v] : -INFINITY;
            float tmax = fmaxf(s0, s1);
            #pragma unroll
            for (int ms = 1; ms < 16; ms <<= 1)
                tmax = fmaxf(tmax, __shfl_xor(tmax, ms, 32));
            const float mnew = fmaxf(mrun[v], tmax);
            const float p0   = __expf(s0 - mnew);
            const float p1   = __expf(s1 - mnew);
            const float corr = __expf(mrun[v] - mnew);
            float rs = p0 + p1;
            #pragma unroll
            for (int ms = 1; ms < 16; ms <<= 1)
                rs += __shfl_xor(rs, ms, 32);
            lrun[v] = lrun[v] * corr + rs;
            mrun[v] = mnew;
            #pragma unroll
            for (int n = 0; n < 4; ++n) oacc[n][v] *= corr;
            // C-layout -> LDS (row-major 16x32 bf16) for A-layout reload
            sP[wave][8 * half + v][lo]      = f32_bf16(p0);
            sP[wave][8 * half + v][16 + lo] = f32_bf16(p1);
        }

        asm volatile("s_wait_dscnt 0x0" ::: "memory"); // LDS in-order per wave; order the reload

        // ---- P as A fragment: row = lo, e<8: col=half*8+e ; e>=8: col=16+half*8+(e-8)
        Frag16 pf;
        {
            const v4u* pp = reinterpret_cast<const v4u*>(&sP[wave][lo][0]);
            pf.q[0] = pp[half];
            pf.q[1] = pp[2 + half];
        }

        // ---- O += P V : V B-frag (e<8: key_local = half*8 + e ; e>=8: 16 + half*8 + (e-8), N = hcol) ----
        #pragma unroll
        for (int n2 = 0; n2 < 4; ++n2) {
            const int hcol = n2 * 16 + lo;
#if HAS_ASYNC_LDS
            const v4u* vp = reinterpret_cast<const v4u*>(vbuf + hcol * 32);
#else
            const v4u* vp = reinterpret_cast<const v4u*>(
                Vt + (size_t)(b * DH + hcol) * TT + jt * 32);
#endif
            Frag16 vf;
            vf.q[0] = vp[half];
            vf.q[1] = vp[2 + half];
            oacc[n2] = wmma_bf16(pf, vf, oacc[n2]);
        }
    }

    // ---- normalize and store f32 output ----
    #pragma unroll
    for (int n2 = 0; n2 < 4; ++n2) {
        #pragma unroll
        for (int v = 0; v < 8; ++v) {
            const int rrow = q0 + 8 * half + v;
            out[((size_t)(b * TT) + rrow) * DH + n2 * 16 + lo] = oacc[n2][v] / lrun[v];
        }
    }
}

// ---------------------------------------------------------------------------
extern "C" void kernel_launch(void* const* d_in, const int* in_sizes, int n_in,
                              void* d_out, int out_size, void* d_ws, size_t ws_size,
                              hipStream_t stream) {
    // setup_inputs order: x, Wk, Wq, Wv
    const float* x  = (const float*)d_in[0];
    const float* Wk = (const float*)d_in[1];
    const float* Wq = (const float*)d_in[2];
    const float* Wv = (const float*)d_in[3];
    float* out = (float*)d_out;

    unsigned short* wsQ  = (unsigned short*)d_ws;            // [B*T, 64] bf16 (pre-scaled)
    unsigned short* wsK  = wsQ + (size_t)BB * TT * DH;       // [B*T, 64] bf16
    unsigned short* wsVt = wsK + (size_t)BB * TT * DH;       // [B*64, T] bf16 (transposed)

    qkv_kernel<<<dim3(BB * TT / 128), dim3(256), 0, stream>>>(
        x, Wq, Wk, Wv, wsQ, wsK, wsVt);
    attn_kernel<<<dim3(BB * TT / 128), dim3(256), 0, stream>>>(
        wsQ, wsK, wsVt, out);
}
